// OptimizedTop2Router_71708773974437
// MI455X (gfx1250) — matrix operand order repaired and checked by reading, not compiled, not run on hardware
//
#include <hip/hip_runtime.h>
#include <hip/hip_bf16.h>

typedef __attribute__((ext_vector_type(16))) __bf16 v16bf;
typedef __attribute__((ext_vector_type(8)))  float  v8f;
typedef __attribute__((ext_vector_type(4)))  float  f4;
typedef __attribute__((ext_vector_type(2)))  float  f2;

#define N_TOK    16384
#define C_DIM    2048
#define E_EXP    64
#define CAPACITY 640                      // int(1.25 * N * 2 / E)
#define COMB_OFF ((long)N_TOK * E_EXP * 2)        // 2097152
#define AUX_OFF  (2L * N_TOK * E_EXP * 2)         // 4194304
#define Z_OFF    (AUX_OFF + 1)

// ---------------- workspace layout (bytes) ----------------
#define WS_LOGITS   0                                   // N*E*4      = 4 MiB
#define WS_WB       (WS_LOGITS + (size_t)N_TOK*E_EXP*4) // E*C*2      = 256 KiB
#define WS_T2I      (WS_WB + (size_t)E_EXP*C_DIM*2)     // N*2*4      = 128 KiB
#define WS_T2V      (WS_T2I + (size_t)N_TOK*2*4)        // N*2*4
#define WS_KEEP     (WS_T2V + (size_t)N_TOK*2*4)        // N*2*4
#define WS_MXSE     (WS_KEEP + (size_t)N_TOK*2*4)       // N*2*4
#define WS_ME       (WS_MXSE + (size_t)N_TOK*2*4)       // 64*4
#define WS_CE       (WS_ME + 64*4)                      // 64*4
#define WS_ZSUM     (WS_CE + 64*4)                      // 4

// Zero the atomically-accumulated scalars each call (harness does not re-poison).
__global__ __launch_bounds__(256) void zero_accum(float* ce, float* zsum) {
    int t = threadIdx.x;
    if (t < 64) ce[t] = 0.0f;
    if (t == 64) zsum[0] = 0.0f;
}

// W (fp32, [E, C]) -> bf16 once; 256 KB result stays resident in L2/WGP$.
__global__ __launch_bounds__(256) void wcvt(const float* __restrict__ W,
                                            __bf16* __restrict__ wb) {
    int i = blockIdx.x * 256 + threadIdx.x;   // E*C = 131072 elements
    wb[i] = (__bf16)W[i];
}

// ---------------------------------------------------------------------------
// Gate GEMM: logits[N, 64] = x[N, 2048] @ W^T, bf16 WMMA with fp32 accumulate.
// Blocking: one wave owns 32 tokens x 64 experts (2 A tiles x 4 B tiles = 8
// accumulators). Each B fragment is reused by both A tiles, halving W/L2
// traffic vs 16-token blocking (~256 MB total, ~43 TB/s demand over the
// ~5.8 us HBM-bound runtime for the 134 MB x stream). 512 waves in flight.
// ---------------------------------------------------------------------------
__device__ __forceinline__ v16bf cvt_a(const float* p) {
    // A fragment slice for this lane: elements 0..15 = K(ka..ka+15),
    // contiguous, matching the B fragment k order.
    f4 a0 = *(const f4*)(p);
    f4 a1 = *(const f4*)(p + 4);
    f4 a2 = *(const f4*)(p + 8);
    f4 a3 = *(const f4*)(p + 12);
    v16bf A;
    A[0]=(__bf16)a0[0]; A[1]=(__bf16)a0[1]; A[2]=(__bf16)a0[2]; A[3]=(__bf16)a0[3];
    A[4]=(__bf16)a1[0]; A[5]=(__bf16)a1[1]; A[6]=(__bf16)a1[2]; A[7]=(__bf16)a1[3];
    A[8]=(__bf16)a2[0]; A[9]=(__bf16)a2[1]; A[10]=(__bf16)a2[2]; A[11]=(__bf16)a2[3];
    A[12]=(__bf16)a3[0]; A[13]=(__bf16)a3[1]; A[14]=(__bf16)a3[2]; A[15]=(__bf16)a3[3];
    return A;
}

__global__ __launch_bounds__(128) void gate_gemm(const float* __restrict__ x,
                                                 const __bf16* __restrict__ wb,
                                                 float* __restrict__ logits) {
    const int lane = threadIdx.x & 31;
    const int wave = threadIdx.x >> 5;
    const int tile = (blockIdx.x * 4 + wave) * 2;  // first of two 16-token tiles
    const int half = lane >> 4;                    // lane group per WMMA layout
    const int l    = lane & 15;

    const float*  xr0 = x + (long)(tile * 16 + l) * C_DIM;   // A tile 0 row
    const float*  xr1 = xr0 + 16L * C_DIM;                   // A tile 1 row
    const __bf16* w0  = wb + (long)( 0 + l) * C_DIM;
    const __bf16* w1  = wb + (long)(16 + l) * C_DIM;
    const __bf16* w2  = wb + (long)(32 + l) * C_DIM;
    const __bf16* w3  = wb + (long)(48 + l) * C_DIM;

    v8f p0 = {}, p1 = {}, p2 = {}, p3 = {};   // token tile 0, expert tiles 0..3
    v8f q0 = {}, q1 = {}, q2 = {}, q3 = {};   // token tile 1, expert tiles 0..3

    for (int k0 = 0; k0 < C_DIM; k0 += 32) {
        __builtin_prefetch(xr0 + k0 + 64, 0, 3);   // WGP-scope global_prefetch
        __builtin_prefetch(xr1 + k0 + 64, 0, 3);

        const int ka = k0 + half * 16;
        v16bf A0 = cvt_a(xr0 + ka);
        v16bf A1 = cvt_a(xr1 + ka);

        // B fragment: contiguous k order (this code's choice, not AMD's table). Lane n holds col n; lanes 0-15
        // carry K 0..15, lanes 16-31 carry K 16..31 -> 16 contiguous bf16.
        const int kb = k0 + half * 16;
        v16bf B0 = *(const v16bf*)(w0 + kb);
        v16bf B1 = *(const v16bf*)(w1 + kb);
        v16bf B2 = *(const v16bf*)(w2 + kb);
        v16bf B3 = *(const v16bf*)(w3 + kb);

        p0 = __builtin_amdgcn_wmma_f32_16x16x32_bf16(false, A0, false, B0, (short)0, p0, false, false);
        q0 = __builtin_amdgcn_wmma_f32_16x16x32_bf16(false, A1, false, B0, (short)0, q0, false, false);
        p1 = __builtin_amdgcn_wmma_f32_16x16x32_bf16(false, A0, false, B1, (short)0, p1, false, false);
        q1 = __builtin_amdgcn_wmma_f32_16x16x32_bf16(false, A1, false, B1, (short)0, q1, false, false);
        p2 = __builtin_amdgcn_wmma_f32_16x16x32_bf16(false, A0, false, B2, (short)0, p2, false, false);
        q2 = __builtin_amdgcn_wmma_f32_16x16x32_bf16(false, A1, false, B2, (short)0, q2, false, false);
        p3 = __builtin_amdgcn_wmma_f32_16x16x32_bf16(false, A0, false, B3, (short)0, p3, false, false);
        q3 = __builtin_amdgcn_wmma_f32_16x16x32_bf16(false, A1, false, B3, (short)0, q3, false, false);
    }

    // C/D layout: lanes 0-15 VGPR j -> M=j; lanes 16-31 VGPR j -> M=8+j; N=l.
    float* o0 = logits + (long)tile * 16 * E_EXP;
    float* o1 = o0 + 16L * E_EXP;
    #pragma unroll
    for (int j = 0; j < 8; ++j) {
        float* oa = o0 + (half * 8 + j) * E_EXP + l;
        oa[0]  = p0[j];
        oa[16] = p1[j];
        oa[32] = p2[j];
        oa[48] = p3[j];
        float* ob = o1 + (half * 8 + j) * E_EXP + l;
        ob[0]  = q0[j];
        ob[16] = q1[j];
        ob[32] = q2[j];
        ob[48] = q3[j];
    }
}

// Per-token softmax statistics, top-2, ce counts (order-free), z-loss partial.
__global__ __launch_bounds__(256) void softmax_top2(const float* __restrict__ logits,
                                                    int* __restrict__ t2i,
                                                    float* __restrict__ t2v,
                                                    float* __restrict__ mx_se,
                                                    float* __restrict__ ce,
                                                    float* __restrict__ zsum) {
    const int n = blockIdx.x * 256 + threadIdx.x;
    const float* row = logits + (long)n * E_EXP;
    float v[E_EXP];
    #pragma unroll
    for (int i = 0; i < E_EXP / 4; ++i) {
        f4 t = *(const f4*)(row + i * 4);
        v[i*4+0] = t[0]; v[i*4+1] = t[1]; v[i*4+2] = t[2]; v[i*4+3] = t[3];
    }
    float mx = v[0], ss = 0.0f;
    #pragma unroll
    for (int e = 0; e < E_EXP; ++e) { mx = fmaxf(mx, v[e]); ss += v[e] * v[e]; }

    float se = 0.0f, b1 = -3.4e38f, b2 = -3.4e38f; int i1 = 0, i2 = 0;
    #pragma unroll
    for (int e = 0; e < E_EXP; ++e) {
        se += __expf(v[e] - mx);
        if (v[e] > b1)      { b2 = b1; i2 = i1; b1 = v[e]; i1 = e; }
        else if (v[e] > b2) { b2 = v[e]; i2 = e; }
    }
    const float inv = 1.0f / se;
    t2i[n*2]     = i1;
    t2i[n*2 + 1] = i2;
    t2v[n*2]     = __expf(b1 - mx) * inv;
    t2v[n*2 + 1] = __expf(b2 - mx) * inv;
    mx_se[n*2]     = mx;
    mx_se[n*2 + 1] = se;
    atomicAdd(&ce[i1], 1.0f);

    __shared__ float red[256];
    red[threadIdx.x] = ss;
    __syncthreads();
    for (int s = 128; s > 0; s >>= 1) {
        if (threadIdx.x < s) red[threadIdx.x] += red[threadIdx.x + s];
        __syncthreads();
    }
    if (threadIdx.x == 0) atomicAdd(zsum, red[0]);
}

// me[e] = mean over tokens of softmax prob for expert e (column reduction).
__global__ __launch_bounds__(256) void me_kernel(const float* __restrict__ logits,
                                                 const float* __restrict__ mx_se,
                                                 float* __restrict__ me) {
    const int e = blockIdx.x;
    float s = 0.0f;
    for (int n = threadIdx.x; n < N_TOK; n += 256)
        s += __expf(logits[(long)n * E_EXP + e] - mx_se[n*2]) / mx_se[n*2 + 1];
    __shared__ float red[256];
    red[threadIdx.x] = s;
    __syncthreads();
    for (int st = 128; st > 0; st >>= 1) {
        if (threadIdx.x < st) red[threadIdx.x] += red[threadIdx.x + st];
        __syncthreads();
    }
    if (threadIdx.x == 0) me[e] = red[0] * (1.0f / N_TOK);
}

// Deterministic token-order exclusive rank per (expert, k): one wave per pair,
// chunked two-pass prefix count (no wave-width-dependent intrinsics).
__global__ __launch_bounds__(32) void positions(const int* __restrict__ t2i,
                                                int* __restrict__ keep) {
    const int e = blockIdx.x >> 1;
    const int k = blockIdx.x & 1;
    const int lane = threadIdx.x;
    const int CH = N_TOK / 32;        // 512 tokens per lane chunk
    __shared__ int cnt[32];

    int c = 0;
    for (int i = 0; i < CH; ++i)
        c += (t2i[(lane * CH + i) * 2 + k] == e);
    cnt[lane] = c;
    __syncthreads();
    int base = 0;
    for (int i = 0; i < lane; ++i) base += cnt[i];

    int pos = base;                   // exclusive rank within expert e, slot k
    for (int i = 0; i < CH; ++i) {
        const int n = lane * CH + i;
        if (t2i[n*2 + k] == e) {
            keep[n*2 + k] = (pos < CAPACITY) ? 1 : 0;
            ++pos;
        }
    }
}

// Scatter into dense [N, E, 2] dispatch_mask + combine_weights (float2 per thread).
__global__ __launch_bounds__(256) void assemble(const int* __restrict__ t2i,
                                                const float* __restrict__ t2v,
                                                const int* __restrict__ keep,
                                                float* __restrict__ out) {
    const long t = (long)blockIdx.x * 256 + threadIdx.x;   // t = n*64 + e
    const int n = (int)(t >> 6);
    const int e = (int)(t & 63);
    const int i0 = t2i[n*2], i1 = t2i[n*2 + 1];
    const float d0 = (e == i0 && keep[n*2])     ? 1.0f : 0.0f;
    const float d1 = (e == i1 && keep[n*2 + 1]) ? 1.0f : 0.0f;
    f2 d; d[0] = d0; d[1] = d1;
    f2 c; c[0] = d0 * t2v[n*2]; c[1] = d1 * t2v[n*2 + 1];
    *(f2*)(out + t*2) = d;
    *(f2*)(out + COMB_OFF + t*2) = c;
}

__global__ void finalize(const float* __restrict__ me, const float* __restrict__ ce,
                         const float* __restrict__ zsum, float* __restrict__ out) {
    if (threadIdx.x == 0) {
        float s = 0.0f;
        for (int e = 0; e < E_EXP; ++e)
            s += me[e] * (ce[e] * (1.0f / N_TOK));
        out[AUX_OFF] = (float)E_EXP * s;
        out[Z_OFF]   = zsum[0] * (1.0f / ((float)N_TOK * E_EXP));
    }
}

extern "C" void kernel_launch(void* const* d_in, const int* in_sizes, int n_in,
                              void* d_out, int out_size, void* d_ws, size_t ws_size,
                              hipStream_t stream) {
    const float* x = (const float*)d_in[0];   // [4, 4096, 2048] fp32
    const float* W = (const float*)d_in[1];   // [64, 2048] fp32
    float* out = (float*)d_out;

    char* ws = (char*)d_ws;
    float*  logits = (float*) (ws + WS_LOGITS);
    __bf16* wb     = (__bf16*)(ws + WS_WB);
    int*    t2i    = (int*)   (ws + WS_T2I);
    float*  t2v    = (float*) (ws + WS_T2V);
    int*    keep   = (int*)   (ws + WS_KEEP);
    float*  mx_se  = (float*) (ws + WS_MXSE);
    float*  me     = (float*) (ws + WS_ME);
    float*  ce     = (float*) (ws + WS_CE);
    float*  zsum   = (float*) (ws + WS_ZSUM);

    zero_accum  <<<1, 256, 0, stream>>>(ce, zsum);
    wcvt        <<<(E_EXP * C_DIM) / 256, 256, 0, stream>>>(W, wb);
    gate_gemm   <<<N_TOK / 32 / 4, 128, 0, stream>>>(x, wb, logits);   // 128 blocks, 512 waves
    softmax_top2<<<N_TOK / 256, 256, 0, stream>>>(logits, t2i, t2v, mx_se, ce, zsum);
    me_kernel   <<<E_EXP, 256, 0, stream>>>(logits, mx_se, me);
    positions   <<<E_EXP * 2, 32, 0, stream>>>(t2i, keep);
    assemble    <<<(N_TOK * E_EXP) / 256, 256, 0, stream>>>(t2i, t2v, keep, out);
    finalize    <<<1, 1, 0, stream>>>(me, ce, zsum, out);
}
